// RSA_Block_48550310314580
// MI455X (gfx1250) — matrix operand order repaired and checked by reading, not compiled, not run on hardware
//
#include <hip/hip_runtime.h>
#include <hip/hip_bf16.h>
#include <cstdint>

// Problem constants (from reference): B=2, C=32, D=16, H=W=64, CK=4
#define BB   2
#define CC   32
#define DD   16
#define NN   4096          // H*W  (sequence length)
#define CKK  4
#define EE   64            // CK*D  (qk dim)
#define MM   512           // C*D   (v dim)

#if __has_builtin(__builtin_amdgcn_sched_barrier)
#define SCHED_FENCE() __builtin_amdgcn_sched_barrier(0)
#else
#define SCHED_FENCE()
#endif

typedef __attribute__((ext_vector_type(16))) _Float16 v16h;
typedef __attribute__((ext_vector_type(8)))  float    v8f;
typedef __attribute__((ext_vector_type(2)))  __fp16   h2;

union F16Frag { v16h h; uint32_t u[8]; uint4 q[2]; };
union H2U { h2 h; int i; };

static __device__ __forceinline__ v8f wmma_f16(const F16Frag& a, const F16Frag& b, v8f c) {
    return __builtin_amdgcn_wmma_f32_16x16x32_f16(false, a.h, false, b.h,
                                                  (short)0, c, false, false);
}

// ---------------------------------------------------------------------------
// Kernel 1: 1x1x1 conv projections -> f16 K/Q/V, laid out so every WMMA
// fragment is two contiguous 16B loads per lane:
//   Kl: [B][N][64]   f16  (A-operand: 16 consecutive e per lane run)
//   Ql: [B][N][64]   f16  (B-operand for S: e contiguous per j)
//   Vm: [B][512][N]  f16  (B-operand for PV: j contiguous per m)
// ---------------------------------------------------------------------------
__global__ __launch_bounds__(256) void proj_kernel(
    const float* __restrict__ x,
    const float* __restrict__ Wk, const float* __restrict__ bk,
    const float* __restrict__ Wq, const float* __restrict__ bq,
    const float* __restrict__ Wv, const float* __restrict__ bv,
    _Float16* __restrict__ Kl, _Float16* __restrict__ Ql,
    _Float16* __restrict__ Vm)
{
    __shared__ float sWk[CKK * CC], sWq[CKK * CC], sWv[CC * CC];
    __shared__ float sbk[CKK], sbq[CKK], sbv[CC];
    for (int t = threadIdx.x; t < CKK * CC; t += blockDim.x) { sWk[t] = Wk[t]; sWq[t] = Wq[t]; }
    for (int t = threadIdx.x; t < CC * CC;  t += blockDim.x) { sWv[t] = Wv[t]; }
    if (threadIdx.x < CKK) { sbk[threadIdx.x] = bk[threadIdx.x]; sbq[threadIdx.x] = bq[threadIdx.x]; }
    if (threadIdx.x < CC)  { sbv[threadIdx.x] = bv[threadIdx.x]; }
    __syncthreads();

    const int tid = blockIdx.x * blockDim.x + threadIdx.x;   // B*D*N threads
    const int j = tid & (NN - 1);
    const int d = (tid >> 12) & (DD - 1);
    const int b = tid >> 16;

    const float* xp = x + ((size_t)(b * CC) * DD + d) * NN + j;   // c-stride = DD*NN

    float ak[CKK], aq[CKK], av[CC];
#pragma unroll
    for (int o = 0; o < CKK; ++o) { ak[o] = sbk[o]; aq[o] = sbq[o]; }
#pragma unroll
    for (int o = 0; o < CC; ++o)  { av[o] = sbv[o]; }

    for (int c = 0; c < CC; ++c) {
        const float xv = xp[(size_t)c * DD * NN];
#pragma unroll
        for (int o = 0; o < CKK; ++o) {
            ak[o] = fmaf(sWk[o * CC + c], xv, ak[o]);
            aq[o] = fmaf(sWq[o * CC + c], xv, aq[o]);
        }
#pragma unroll
        for (int o = 0; o < CC; ++o) av[o] = fmaf(sWv[o * CC + c], xv, av[o]);
    }

    // K, Q: [b][i=j][e = ck*16+d]
    const size_t kq = ((size_t)b * NN + j) * EE;
#pragma unroll
    for (int ck = 0; ck < CKK; ++ck) {
        Kl[kq + ck * 16 + d] = (_Float16)ak[ck];
        Ql[kq + ck * 16 + d] = (_Float16)aq[ck];
    }
    // V: [b][m = c*16+d][j]
#pragma unroll
    for (int c = 0; c < CC; ++c) {
        const int m = c * 16 + d;
        Vm[((size_t)b * MM + m) * NN + j] = (_Float16)av[c];
    }
}

// ---------------------------------------------------------------------------
// Kernel 2: fused flash-attention, 64-wide j-blocks, b128 fragments.
// sched_barrier-pinned load phases + distance-2 pipelined V fragments.
// Grid: B * (N/32) blocks of 256 threads (8 waves).
// Wave w: i-tile t = w>>2 (16 queries), d_v slice s = w&3 (128 of 512).
// ---------------------------------------------------------------------------
__global__ __launch_bounds__(256) void attn_kernel(
    const float* __restrict__ x,
    const _Float16* __restrict__ Kl,
    const _Float16* __restrict__ Ql,
    const _Float16* __restrict__ Vm,
    const float* __restrict__ gamma,
    float* __restrict__ out)
{
    __shared__ __align__(16) _Float16 pscr[8][16 * 64];   // per-wave P scratch (2KB)

    const int wid  = threadIdx.x >> 5;
    const int lane = threadIdx.x & 31;
    const int g    = lane >> 4;                 // 16-lane half
    const int ln   = lane & 15;
    const int t    = wid >> 2;                  // i-subtile within block
    const int s    = wid & 3;                   // value-dim slice
    const int b    = blockIdx.x >> 7;           // 128 query-blocks per batch
    const int iw   = ((blockIdx.x & 127) << 5) + t * 16;   // wave's 16-query base
    const int mb   = s * 128;

    // --- persistent A-operand: K rows for this wave's 16 queries (e = 0..63) ---
    F16Frag aK[2];
    {
        const uint4* kp4 = (const uint4*)(Kl + ((size_t)b * NN + iw + ln) * EE);
#pragma unroll
        for (int ec = 0; ec < 2; ++ec) {
            aK[ec].q[0] = kp4[ec * 4 + 2 * g];
            aK[ec].q[1] = kp4[ec * 4 + 2 * g + 1];
        }
    }

    // all-ones B operand for WMMA-based row sums
    F16Frag ones;
#pragma unroll
    for (int qq = 0; qq < 16; ++qq) ones.h[qq] = (_Float16)1.0f;

    v8f   O[8];
    float mrow[8], lrow[8];
#pragma unroll
    for (int mt = 0; mt < 8; ++mt) O[mt] = (v8f){};
#pragma unroll
    for (int v = 0; v < 8; ++v) { mrow[v] = -INFINITY; lrow[v] = 0.0f; }

    const _Float16* qrow0 = Ql + (size_t)b * NN * EE;   // rows of 64 f16 (8 uint4)
    const _Float16* vrow0 = Vm + (size_t)b * MM * NN;   // rows of 4096 f16 (512 uint4)
    _Float16* pl = pscr[wid];

    for (int j0 = 0; j0 < NN; j0 += 64) {
        // prefetch next j-block of V rows for this wave's m-slice
        if (j0 + 64 < NN)
            __builtin_prefetch((const void*)(vrow0 + ((size_t)(mb + lane * 4)) * NN + j0 + 64), 0, 3);

        // ----- preload all 8 Q fragments, then fence: loads issue as one clause,
        //       waits in the WMMA phase become staggered instead of loadcnt==0 -----
        F16Frag bq[4][2];
#pragma unroll
        for (int jt = 0; jt < 4; ++jt) {
            const uint4* qp4 = (const uint4*)(qrow0 + (size_t)(j0 + jt * 16 + ln) * EE);
            bq[jt][0].q[0] = qp4[g * 2];          bq[jt][0].q[1] = qp4[g * 2 + 1];
            bq[jt][1].q[0] = qp4[4 + g * 2];      bq[jt][1].q[1] = qp4[4 + g * 2 + 1];
        }
        SCHED_FENCE();

        // ----- S = K_i^T Q_j : four 16x16 D-tiles covering 64 j's -----
        v8f S[4];
#pragma unroll
        for (int jt = 0; jt < 4; ++jt) {
            v8f acc = (v8f){};
            acc   = wmma_f16(aK[0], bq[jt][0], acc);
            S[jt] = wmma_f16(aK[1], bq[jt][1], acc);
        }

        // ----- online softmax: row-max with packed-f16 butterflies -----
        float mx[8];
#pragma unroll
        for (int v = 0; v < 8; ++v)
            mx[v] = fmaxf(fmaxf(S[0][v], S[1][v]), fmaxf(S[2][v], S[3][v]));
        H2U pk[4];
#pragma unroll
        for (int k = 0; k < 4; ++k)
            pk[k].h = __builtin_amdgcn_cvt_pkrtz(mx[2 * k], mx[2 * k + 1]);
#pragma unroll
        for (int o = 1; o < 16; o <<= 1)
#pragma unroll
            for (int k = 0; k < 4; ++k) {
                H2U tt; tt.i = __shfl_xor(pk[k].i, o, 32);
                pk[k].h = __builtin_elementwise_max(pk[k].h, tt.h);
            }

        float alpha[8];
#pragma unroll
        for (int v = 0; v < 8; ++v) {
            const float mnew = fmaxf(mrow[v], (float)pk[v >> 1].h[v & 1]);
            alpha[v] = __expf(mrow[v] - mnew);
            mrow[v]  = mnew;
            // P to LDS in (i, j) row-major (row = v+8g, cols jt*16+ln)
#pragma unroll
            for (int jt = 0; jt < 4; ++jt)
                pl[(v + 8 * g) * 64 + jt * 16 + ln] = (_Float16)__expf(S[jt][v] - mnew);
            // rescale accumulators for this row
#pragma unroll
            for (int mt = 0; mt < 8; ++mt) O[mt][v] *= alpha[v];
        }
        asm volatile("s_wait_dscnt 0" ::: "memory");   // wave-private LDS RAW fence

        // ----- reload P as two A-operands (i x 32j each), 4x ds_load_b128 -----
        F16Frag aP[2];
        {
            const uint4* plu4 = (const uint4*)pl;   // row = 8 uint4
#pragma unroll
            for (int c2 = 0; c2 < 2; ++c2) {
                aP[c2].q[0] = plu4[ln * 8 + c2 * 4 + 2 * g];
                aP[c2].q[1] = plu4[ln * 8 + c2 * 4 + 2 * g + 1];
            }
        }

        // ----- row sums via WMMA: rs[v] = sum_j P[row v+8g][j] -----
        {
            v8f rs = (v8f){};
            rs = wmma_f16(aP[0], ones, rs);
            rs = wmma_f16(aP[1], ones, rs);
#pragma unroll
            for (int v = 0; v < 8; ++v) lrow[v] = lrow[v] * alpha[v] + rs[v];
        }

        // ----- O += P * V : distance-2 pipelined V fragments (3 buffers) -----
        F16Frag bV[3][2];
#pragma unroll
        for (int p0 = 0; p0 < 2; ++p0) {
            const uint4* vp4 = (const uint4*)(vrow0 + (size_t)(mb + p0 * 16 + ln) * NN) + (j0 >> 3);
            bV[p0][0].q[0] = vp4[g * 2];       bV[p0][0].q[1] = vp4[g * 2 + 1];
            bV[p0][1].q[0] = vp4[4 + g * 2];   bV[p0][1].q[1] = vp4[4 + g * 2 + 1];
        }
#pragma unroll
        for (int mt = 0; mt < 8; ++mt) {
            const int cur = mt % 3, wr = (mt + 2) % 3;
            if (mt < 6) {
                const uint4* vp4 = (const uint4*)(vrow0 + (size_t)(mb + (mt + 2) * 16 + ln) * NN) + (j0 >> 3);
                bV[wr][0].q[0] = vp4[g * 2];       bV[wr][0].q[1] = vp4[g * 2 + 1];
                bV[wr][1].q[0] = vp4[4 + g * 2];   bV[wr][1].q[1] = vp4[4 + g * 2 + 1];
            }
            SCHED_FENCE();   // keep look-ahead loads issued before consuming WMMAs
            O[mt] = wmma_f16(aP[0], bV[cur][0], O[mt]);
            O[mt] = wmma_f16(aP[1], bV[cur][1], O[mt]);
        }
    }

    // ----- epilogue: out = gamma*x + O/l  (vectorized: 8 consecutive i per lane) -----
    const float gm = gamma[0];
    float rl[8];
#pragma unroll
    for (int v = 0; v < 8; ++v) rl[v] = 1.0f / lrow[v];
#pragma unroll
    for (int mt = 0; mt < 8; ++mt) {
        const int m = mb + mt * 16 + ln;                  // m = c*16 + d
        const size_t base = ((size_t)b * MM + m) * NN + iw + 8 * g;
        const float4* xin = (const float4*)(x + base);
        float4*       op  = (float4*)(out + base);
        float4 r0, r1;
        r0.x = fmaf(gm, xin[0].x, O[mt][0] * rl[0]);
        r0.y = fmaf(gm, xin[0].y, O[mt][1] * rl[1]);
        r0.z = fmaf(gm, xin[0].z, O[mt][2] * rl[2]);
        r0.w = fmaf(gm, xin[0].w, O[mt][3] * rl[3]);
        r1.x = fmaf(gm, xin[1].x, O[mt][4] * rl[4]);
        r1.y = fmaf(gm, xin[1].y, O[mt][5] * rl[5]);
        r1.z = fmaf(gm, xin[1].z, O[mt][6] * rl[6]);
        r1.w = fmaf(gm, xin[1].w, O[mt][7] * rl[7]);
        op[0] = r0;
        op[1] = r1;
    }
}

// ---------------------------------------------------------------------------
extern "C" void kernel_launch(void* const* d_in, const int* in_sizes, int n_in,
                              void* d_out, int out_size, void* d_ws, size_t ws_size,
                              hipStream_t stream) {
    const float* x     = (const float*)d_in[0];
    const float* Wk    = (const float*)d_in[1];
    const float* bk    = (const float*)d_in[2];
    const float* Wq    = (const float*)d_in[3];
    const float* bq    = (const float*)d_in[4];
    const float* Wv    = (const float*)d_in[5];
    const float* bv    = (const float*)d_in[6];
    const float* gamma = (const float*)d_in[7];
    float* out = (float*)d_out;

    // workspace layout (f16): K 1MB | Q 1MB | V 8MB
    _Float16* Kl = (_Float16*)d_ws;                              // B*N*64
    _Float16* Ql = Kl + (size_t)BB * NN * EE;                    // B*N*64
    _Float16* Vm = Ql + (size_t)BB * NN * EE;                    // B*512*N

    const int projThreads = BB * DD * NN;                        // 131072
    proj_kernel<<<projThreads / 256, 256, 0, stream>>>(x, Wk, bk, Wq, bq, Wv, bv,
                                                       Kl, Ql, Vm);
    attn_kernel<<<BB * (NN / 32), 256, 0, stream>>>(x, Kl, Ql, Vm, gamma, out);
}
